// SimplicialHopfieldNetwork_58110907515217
// MI455X (gfx1250) — matrix operand order repaired and checked by reading, not compiled, not run on hardware
//
#include <hip/hip_runtime.h>
#include <hip/hip_bf16.h>

typedef __attribute__((ext_vector_type(16))) __bf16 v16bf;
typedef __attribute__((ext_vector_type(8)))  float  v8f;
typedef unsigned short u16;
typedef unsigned int   u32;

// ---------------- helpers ----------------

__device__ __forceinline__ u16 f32_to_bf16(float f) {
    u32 u = __float_as_uint(f);
    u32 lsb = (u >> 16) & 1u;
    u += 0x7FFFu + lsb;               // round-to-nearest-even
    return (u16)(u >> 16);
}

// ---------------- kernel 0: zero accumulators ----------------

__global__ void k_init(float* accum) {
    accum[0] = 0.0f;   // total LSE sum
    accum[1] = 0.0f;   // sum(g*g)
}

// ---------------- kernel 1: f32 -> bf16 conversion + sum(g*g) ----------------
// gbf[m][d] = bf16(g[m][d])            (row-major, M = B*N rows, D cols)
// pT [k][d] = bf16(patterns[d][k])     (transposed so B-fragment loads are contiguous)

__global__ void k_prep(const float* __restrict__ g, const float* __restrict__ pat,
                       u16* __restrict__ gbf, u16* __restrict__ pT,
                       int nG, int nP, int K, int D, float* __restrict__ accum) {
    int i = blockIdx.x * blockDim.x + threadIdx.x;
    float sq = 0.0f;
    if (i < nG) {
        float v = g[i];
        gbf[i] = f32_to_bf16(v);
        sq = v * v;
    }
    if (i < nP) {
        int d = i / K;
        int k = i - d * K;
        pT[k * D + d] = f32_to_bf16(pat[i]);
    }
    // wave reduce then block reduce sum(g*g)
    for (int off = 16; off > 0; off >>= 1)
        sq += __shfl_xor(sq, off, 32);
    __shared__ float ssum;
    if (threadIdx.x == 0) ssum = 0.0f;
    __syncthreads();
    if ((threadIdx.x & 31) == 0) atomicAdd(&ssum, sq);
    __syncthreads();
    if (threadIdx.x == 0) atomicAdd(&accum[1], ssum);
}

// ---------------- kernel 2: WMMA GEMM  h[M][K] = gbf[M][D] * pT[K][D]^T ----------------
// One wave per 16x16 output tile; 8 x v_wmma_f32_16x16x32_bf16 over D=256.

__global__ void k_gemm_wmma(const u16* __restrict__ gbf, const u16* __restrict__ pT,
                            float* __restrict__ h, int M, int K, int D) {
    const int lane  = threadIdx.x & 31;
    const int wave  = (blockIdx.x * blockDim.x + threadIdx.x) >> 5;
    const int tilesN = K >> 4;                 // 64
    const int tM = wave / tilesN;              // 0..M/16-1
    const int tN = wave - tM * tilesN;         // 0..63
    if (tM >= (M >> 4)) return;

    const int half = lane >> 4;                // 0 or 1
    const int l16  = lane & 15;

    const u16* aRow = gbf + (tM * 16 + l16) * D;   // A: lane holds M-row l16
    const u16* bRow = pT  + (tN * 16 + l16) * D;   // B: lane holds N-col l16 (contiguous in d)

    v8f acc = {};
    for (int d0 = 0; d0 < D; d0 += 32) {
        union { v16bf v; u32 u[8]; } A, B;
        #pragma unroll
        for (int v = 0; v < 8; ++v) {
            // A-matrix 16x32 bf16 layout (ISA 7.12.2): VGPR v holds K pair
            int kA = (v < 4 ? 2 * v : 16 + 2 * (v - 4)) + 8 * half;
            A.u[v] = *(const u32*)(aRow + d0 + kA);
            // B-matrix 32x16 bf16 layout: same K order as A
            int kB = (v < 4 ? 2 * v : 16 + 2 * (v - 4)) + 8 * half;
            B.u[v] = *(const u32*)(bRow + d0 + kB);
        }
        acc = __builtin_amdgcn_wmma_f32_16x16x32_bf16(
            /*neg_a=*/false, A.v, /*neg_b=*/false, B.v,
            /*c_mod=*/(short)0, acc, /*reuse_a=*/false, /*reuse_b=*/false);
    }

    // C/D 16x16 f32 layout: VGPR r -> row 8*half + r, col l16
    #pragma unroll
    for (int r = 0; r < 8; ++r) {
        int row = tM * 16 + half * 8 + r;
        int col = tN * 16 + l16;
        h[row * K + col] = acc[r];
    }
}

// ---------------- kernel 3: per-simplex logsumexp, one wave per (batch, simplex) ----------------

__global__ void k_simplex_lse(const float* __restrict__ h,
                              const int* __restrict__ edges, int nE,
                              const int* __restrict__ tris,  int nT,
                              const float* __restrict__ beta_p,
                              int B, int N, int K,
                              float* __restrict__ accum) {
    __shared__ float blockSum;
    if (threadIdx.x == 0) blockSum = 0.0f;
    __syncthreads();

    const int lane = threadIdx.x & 31;
    const int wid  = (blockIdx.x * blockDim.x + threadIdx.x) >> 5;
    const int perB  = nE + nT;
    const int total = B * perB;
    const float beta = *beta_p;
    const int KPL = K >> 5;                    // elements per lane (32)

    if (wid < total) {
        int b = wid / perB;
        int s = wid - b * perB;

        float vals[32];
        float vmax = -3.402823466e38f;

        if (s < nE) {
            const float* r0 = h + (size_t)(b * N + edges[2 * s    ]) * K;
            const float* r1 = h + (size_t)(b * N + edges[2 * s + 1]) * K;
            #pragma unroll
            for (int t = 0; t < 32; ++t) {
                int k = lane + (t < KPL ? t : 0) * 32;
                float v = beta * (r0[k] + r1[k]);
                vals[t] = v;
                vmax = fmaxf(vmax, v);
            }
        } else {
            int tr = s - nE;
            const float* r0 = h + (size_t)(b * N + tris[3 * tr    ]) * K;
            const float* r1 = h + (size_t)(b * N + tris[3 * tr + 1]) * K;
            const float* r2 = h + (size_t)(b * N + tris[3 * tr + 2]) * K;
            #pragma unroll
            for (int t = 0; t < 32; ++t) {
                int k = lane + (t < KPL ? t : 0) * 32;
                float v = beta * (r0[k] + r1[k] + r2[k]);
                vals[t] = v;
                vmax = fmaxf(vmax, v);
            }
        }

        for (int off = 16; off > 0; off >>= 1)
            vmax = fmaxf(vmax, __shfl_xor(vmax, off, 32));

        float se = 0.0f;
        #pragma unroll
        for (int t = 0; t < 32; ++t)
            se += __expf(vals[t] - vmax);
        for (int off = 16; off > 0; off >>= 1)
            se += __shfl_xor(se, off, 32);

        if (lane == 0)
            atomicAdd(&blockSum, __logf(se) + vmax);
    }

    __syncthreads();
    if (threadIdx.x == 0) atomicAdd(&accum[0], blockSum);
}

// ---------------- kernel 4: finalize scalar ----------------

__global__ void k_finalize(const float* __restrict__ accum,
                           const float* __restrict__ beta_p,
                           int numSimplices, int BN, float* __restrict__ out) {
    float lse = accum[0];
    float gg  = accum[1];
    float beta = *beta_p;
    float energy_patterns = -(1.0f / (beta * (float)numSimplices)) * lse;
    float energy_reg = -2.0f * gg;
    out[0] = (energy_patterns + energy_reg) / (float)BN;
}

// ---------------- launch ----------------

extern "C" void kernel_launch(void* const* d_in, const int* in_sizes, int n_in,
                              void* d_out, int out_size, void* d_ws, size_t ws_size,
                              hipStream_t stream) {
    const float* g    = (const float*)d_in[0];
    const float* pat  = (const float*)d_in[1];
    const float* beta = (const float*)d_in[2];
    const int*   edges = (const int*)d_in[3];
    const int*   tris  = (const int*)d_in[4];

    const int nG = in_sizes[0];          // B*N*D = 262144
    const int nP = in_sizes[1];          // D*K   = 262144
    const int nE = in_sizes[3] / 2;      // 16320
    const int nT = in_sizes[4] / 3;      // 16320
    const int K  = 1024;
    const int D  = nP / K;               // 256
    const int M  = nG / D;               // B*N = 1024
    const int N  = 256;
    const int B  = M / N;                // 4

    // workspace layout (all 256B aligned)
    char* ws = (char*)d_ws;
    float* accum = (float*)ws;                                 // 2 floats
    u16*   gbf   = (u16*)(ws + 256);                           // M*D bf16   (512 KB)
    u16*   pT    = (u16*)(ws + 256 + (size_t)nG * 2);          // K*D bf16   (512 KB)
    float* h     = (float*)(ws + 256 + (size_t)nG * 2 + (size_t)nP * 2);  // M*K f32 (4 MB)

    // 0) zero accumulators
    k_init<<<1, 1, 0, stream>>>(accum);

    // 1) bf16 conversion + sum(g*g)
    {
        int n = nG > nP ? nG : nP;
        int blocks = (n + 255) / 256;
        k_prep<<<blocks, 256, 0, stream>>>(g, pat, gbf, pT, nG, nP, K, D, accum);
    }

    // 2) WMMA GEMM: one wave per 16x16 tile, 8 waves per block
    {
        int waves  = (M / 16) * (K / 16);      // 4096
        int blocks = (waves + 7) / 8;          // 512
        k_gemm_wmma<<<blocks, 256, 0, stream>>>(gbf, pT, h, M, K, D);
    }

    // 3) per-simplex LSE: one wave per (batch, simplex)
    {
        int totalWaves = B * (nE + nT);        // 130560
        int blocks = (totalWaves + 7) / 8;     // 16320
        k_simplex_lse<<<blocks, 256, 0, stream>>>(h, edges, nE, tris, nT, beta,
                                                  B, N, K, accum);
    }

    // 4) finalize
    k_finalize<<<1, 1, 0, stream>>>(accum, beta, nE + nT, M, (float*)d_out);
}
